// GQA_69157563400630
// MI455X (gfx1250) — matrix operand order repaired and checked by reading, not compiled, not run on hardware
//
#include <hip/hip_runtime.h>
#include <hip/hip_bf16.h>

typedef __attribute__((ext_vector_type(16))) _Float16 v16h;
typedef __attribute__((ext_vector_type(8)))  _Float16 v8h;
typedef __attribute__((ext_vector_type(4)))  _Float16 v4h;
typedef __attribute__((ext_vector_type(8)))  float    v8f;
typedef __attribute__((ext_vector_type(4)))  int      v4i;

#define B_   2
#define L_   2048
#define D_   2048
#define H_   16
#define HD_  128
#define M_   (B_*L_)     /* 4096 */
#define NQ_  (H_*HD_)    /* 2048 */
#define NKV_ 512
#define SCALE_ 0.08838834764831845f
#define NEG_  -1.0e30f

#if defined(__gfx1250__) && \
    __has_builtin(__builtin_amdgcn_global_load_async_to_lds_b128) && \
    __has_builtin(__builtin_amdgcn_s_wait_asynccnt)
#define USE_ASYNC_LDS 1
#else
#define USE_ASYNC_LDS 0
#endif

static __device__ __forceinline__ v16h cat16(v8h a, v8h b) {
  return __builtin_shufflevector(a, b, 0,1,2,3,4,5,6,7,8,9,10,11,12,13,14,15);
}
static __device__ __forceinline__ v8f wmma16(v16h a, v16h b, v8f c) {
  // D = A(16x32 f16) * B(32x16 f16) + C(16x16 f32)
  return __builtin_amdgcn_wmma_f32_16x16x32_f16(false, a, false, b, (short)0, c,
                                                false, false);
}

__global__ void cvt_f32_f16_v4(const float4* __restrict__ in,
                               _Float16* __restrict__ out, int n4) {
  int i = blockIdx.x * blockDim.x + threadIdx.x;
  if (i < n4) {
    float4 f = in[i];
    v4h h;
    h.x = (_Float16)f.x; h.y = (_Float16)f.y;
    h.z = (_Float16)f.z; h.w = (_Float16)f.w;
    *(v4h*)(out + (size_t)i * 4) = h;
  }
}

// C[M,N] = A[M,K] @ Bw[K,N] + bias ; out f16 or f32 (template). Block tile
// 128x128, 8 waves, wave tile 64(M) x 32(N) = 4x2 WMMA tiles.
template <bool OUT_F32>
__global__ __launch_bounds__(256)
void gemm_f16_wmma(const _Float16* __restrict__ A, const _Float16* __restrict__ Bw,
                   const float* __restrict__ bias, void* __restrict__ Cout,
                   int Ndim, int Kdim) {
  __shared__ __align__(16) _Float16 sA[128 * 32];   // [m][k] row-major
  __shared__ __align__(16) _Float16 sBT[128 * 32];  // [n][k] (B transposed)

  const int tid  = threadIdx.x;
  const int lane = tid & 31;
  const int w    = tid >> 5;
  const int m0   = blockIdx.y * 128;
  const int n0   = blockIdx.x * 128;
  const int m_sub = (w >> 2) * 64;
  const int n_sub = (w & 3) * 32;

  const int rsel = lane & 15;
  const int radd = (lane & 16) ? 8 : 0;
  const int alo  = (lane < 16) ? 0 : 8;   // A-frag K start (half idx)
  const int blo  = (lane < 16) ? 0 : 8;   // B-frag K start

  v8f acc[4][2];
  v8f zero8 = {0.f,0.f,0.f,0.f,0.f,0.f,0.f,0.f};
  #pragma unroll
  for (int i = 0; i < 4; ++i)
    #pragma unroll
    for (int j = 0; j < 2; ++j) acc[i][j] = zero8;

  const int row_a  = tid >> 1;          // 0..127
  const int part_a = (tid & 1) * 16;    // 0/16
  const int k_b    = tid >> 3;          // 0..31
  const int nb_b   = (tid & 7) * 16;    // 0..112

  for (int kt = 0; kt < Kdim; kt += 32) {
    { // stage A tile 128x32 (contiguous copy -> async global->LDS if available)
      const _Float16* src = A + (size_t)(m0 + row_a) * Kdim + kt + part_a;
      _Float16* dst = &sA[row_a * 32 + part_a];
#if USE_ASYNC_LDS
      __builtin_amdgcn_global_load_async_to_lds_b128(
          (v4i*)src, (v4i*)dst, 0, 0);
      __builtin_amdgcn_global_load_async_to_lds_b128(
          (v4i*)(src + 8), (v4i*)(dst + 8), 0, 0);
#else
      *(v8h*)dst       = *(const v8h*)src;
      *(v8h*)(dst + 8) = *(const v8h*)(src + 8);
#endif
    }
    { // stage B tile 32x128, transposed into sBT[n][k]
      const _Float16* src = Bw + (size_t)(kt + k_b) * Ndim + n0 + nb_b;
      union { v8h v[2]; _Float16 e[16]; } u;
      u.v[0] = *(const v8h*)src;
      u.v[1] = *(const v8h*)(src + 8);
      #pragma unroll
      for (int j = 0; j < 16; ++j) sBT[(nb_b + j) * 32 + k_b] = u.e[j];
      // prefetch next k-tile of B into caches (global_prefetch_b8)
      if (kt + 32 < Kdim)
        __builtin_prefetch((const void*)(src + (size_t)32 * Ndim), 0, 1);
    }
#if USE_ASYNC_LDS
    __builtin_amdgcn_s_wait_asynccnt(0);
#endif
    __syncthreads();

    v16h afr[4];
    #pragma unroll
    for (int i = 0; i < 4; ++i) {
      const int row = m_sub + i * 16 + rsel;
      afr[i] = cat16(*(const v8h*)&sA[row * 32 + alo],
                     *(const v8h*)&sA[row * 32 + alo + 16]);
    }
    v16h bfr[2];
    #pragma unroll
    for (int j = 0; j < 2; ++j) {
      const int n = n_sub + j * 16 + rsel;
      bfr[j] = cat16(*(const v8h*)&sBT[n * 32 + blo],
                     *(const v8h*)&sBT[n * 32 + blo + 16]);
    }
    #pragma unroll
    for (int i = 0; i < 4; ++i)
      #pragma unroll
      for (int j = 0; j < 2; ++j)
        acc[i][j] = wmma16(afr[i], bfr[j], acc[i][j]);
    __syncthreads();
  }

  #pragma unroll
  for (int i = 0; i < 4; ++i) {
    #pragma unroll
    for (int j = 0; j < 2; ++j) {
      const int ncol = n0 + n_sub + j * 16 + rsel;
      const float bv = bias[ncol];
      const size_t base = (size_t)(m0 + m_sub + i * 16 + radd) * Ndim + ncol;
      #pragma unroll
      for (int r = 0; r < 8; ++r) {
        const float val = acc[i][j][r] + bv;
        if (OUT_F32) ((float*)Cout)[base + (size_t)r * Ndim] = val;
        else ((_Float16*)Cout)[base + (size_t)r * Ndim] = (_Float16)val;
      }
    }
  }
}

// Flash attention with GQA interleaved-repeat expansion.
// grid (L/128, H, B); 8 waves, wave w owns 16 query rows; 32-key blocks.
__global__ __launch_bounds__(256)
void flash_gqa_wmma(const _Float16* __restrict__ Qf, const _Float16* __restrict__ Kf,
                    const _Float16* __restrict__ Vf, _Float16* __restrict__ Af) {
  __shared__ __align__(16) _Float16 sK[32 * 128];   // [key][d] expanded
  __shared__ __align__(16) _Float16 sVT[128 * 32];  // [d][key] expanded
  __shared__ __align__(16) float    sS[8 * 16 * 32];
  __shared__ __align__(16) _Float16 sP[8 * 16 * 32];
  __shared__ float sC[8 * 16];

  const int tid  = threadIdx.x;
  const int lane = tid & 31;
  const int w    = tid >> 5;
  const int qblk = blockIdx.x;
  const int h    = blockIdx.y;
  const int b    = blockIdx.z;
  const int q0   = qblk * 128 + w * 16;
  const int rsel = lane & 15;
  const int radd = (lane & 16) ? 8 : 0;
  const int alo  = (lane < 16) ? 0 : 8;
  const int blo  = (lane < 16) ? 0 : 8;

  float*    sSw = &sS[w * 16 * 32];
  _Float16* sPw = &sP[w * 16 * 32];
  float*    sCw = &sC[w * 16];

  // Q fragments resident in VGPRs: 16 rows x HD(=128) in 4 K-chunks of 32
  v16h qfr[4];
  {
    const size_t base = ((size_t)(b * L_ + q0 + rsel)) * NQ_ + h * HD_;
    #pragma unroll
    for (int c = 0; c < 4; ++c) {
      const _Float16* p = Qf + base + c * 32 + alo;
      qfr[c] = cat16(*(const v8h*)p, *(const v8h*)(p + 16));
    }
  }

  v8f zero8 = {0.f,0.f,0.f,0.f,0.f,0.f,0.f,0.f};
  v8f o[8];
  #pragma unroll
  for (int t = 0; t < 8; ++t) o[t] = zero8;
  float mrow = NEG_, lrow = 0.f;  // live in lanes 0..15

  const int key_l = tid >> 3;        // 0..31
  const int doff  = (tid & 7) * 16;  // 0..112

  const int nkb = (qblk + 1) * 4;
  for (int kb = 0; kb < nkb; ++kb) {
    const int key0 = kb * 32;
    __syncthreads();
    { // stage K/V with GQA expansion: exp[d] = raw[h*32 + d/4]
      const size_t src = ((size_t)(b * L_ + key0 + key_l)) * NKV_ + h * 32 + (doff >> 2);
      _Float16 kv[4], vv[4];
      #pragma unroll
      for (int u = 0; u < 4; ++u) { kv[u] = Kf[src + u]; vv[u] = Vf[src + u]; }
      #pragma unroll
      for (int j = 0; j < 16; ++j) {
        sK[key_l * 128 + doff + j]   = kv[j >> 2];
        sVT[(doff + j) * 32 + key_l] = vv[j >> 2];
      }
    }
    __syncthreads();

    // S = Q @ K^T   (16 x 32 keys, 2 N-tiles, 4 K-chunks over HD)
    v8f s[2]; s[0] = zero8; s[1] = zero8;
    #pragma unroll
    for (int c = 0; c < 4; ++c) {
      #pragma unroll
      for (int j = 0; j < 2; ++j) {
        const _Float16* p = &sK[(j * 16 + rsel) * 128 + c * 32 + blo];
        v16h bf = cat16(*(const v8h*)p, *(const v8h*)(p + 16));
        s[j] = wmma16(qfr[c], bf, s[j]);
      }
    }
    #pragma unroll
    for (int j = 0; j < 2; ++j)
      #pragma unroll
      for (int r = 0; r < 8; ++r)
        sSw[(r + radd) * 32 + j * 16 + rsel] = s[j][r];
    __syncthreads();

    // online softmax, one row per lane (lanes 0..15)
    if (lane < 16) {
      const int row = lane, qidx = q0 + row;
      float rmax = NEG_;
      #pragma unroll
      for (int kc = 0; kc < 32; ++kc) {
        float sv = (key0 + kc <= qidx) ? sSw[row * 32 + kc] * SCALE_ : NEG_;
        rmax = fmaxf(rmax, sv);
      }
      const float mnew = fmaxf(mrow, rmax);
      const float corr = __expf(mrow - mnew);
      float sum = 0.f;
      #pragma unroll
      for (int kc = 0; kc < 32; ++kc) {
        float sv = (key0 + kc <= qidx) ? sSw[row * 32 + kc] * SCALE_ : NEG_;
        float pv = __expf(sv - mnew);
        sum += pv;
        sPw[row * 32 + kc] = (_Float16)pv;
      }
      lrow = lrow * corr + sum;
      mrow = mnew;
      sCw[row] = corr;
    }
    __syncthreads();

    // O = O*corr + P @ V
    float cr[8];
    #pragma unroll
    for (int r = 0; r < 8; ++r) cr[r] = sCw[r + radd];
    #pragma unroll
    for (int t = 0; t < 8; ++t)
      #pragma unroll
      for (int r = 0; r < 8; ++r)
        o[t][r] = o[t][r] * cr[r];

    const _Float16* pp = &sPw[rsel * 32 + alo];
    v16h pfr = cat16(*(const v8h*)pp, *(const v8h*)(pp + 16));
    #pragma unroll
    for (int t = 0; t < 8; ++t) {
      const _Float16* vp = &sVT[(t * 16 + rsel) * 32 + blo];
      v16h vfb = cat16(*(const v8h*)vp, *(const v8h*)(vp + 16));
      o[t] = wmma16(pfr, vfb, o[t]);
    }
  }

  __syncthreads();
  if (lane < 16) sCw[lane] = 1.0f / lrow;
  __syncthreads();
  #pragma unroll
  for (int r = 0; r < 8; ++r) {
    const float il = sCw[r + radd];
    const size_t base = ((size_t)(b * L_ + q0 + r + radd)) * NQ_ + h * HD_;
    #pragma unroll
    for (int t = 0; t < 8; ++t)
      Af[base + t * 16 + rsel] = (_Float16)(o[t][r] * il);
  }
}

extern "C" void kernel_launch(void* const* d_in, const int* in_sizes, int n_in,
                              void* d_out, int out_size, void* d_ws, size_t ws_size,
                              hipStream_t stream) {
  (void)in_sizes; (void)n_in; (void)out_size; (void)ws_size;
  const float* x  = (const float*)d_in[0];
  const float* Wq = (const float*)d_in[1];
  const float* bq = (const float*)d_in[2];
  const float* Wk = (const float*)d_in[3];
  const float* bk = (const float*)d_in[4];
  const float* Wv = (const float*)d_in[5];
  const float* bv = (const float*)d_in[6];
  const float* Wo = (const float*)d_in[7];
  const float* bo = (const float*)d_in[8];

  char* ws = (char*)d_ws;
  _Float16* xh  = (_Float16*)(ws);              // M*D        f16 = 16 MB
  _Float16* Wqh = (_Float16*)(ws + 16777216);   // D*NQ
  _Float16* Wkh = (_Float16*)(ws + 25165824);   // D*NKV
  _Float16* Wvh = (_Float16*)(ws + 27262976);   // D*NKV
  _Float16* Woh = (_Float16*)(ws + 29360128);   // NQ*D
  _Float16* Qf  = (_Float16*)(ws + 37748736);   // M*NQ
  _Float16* Kf  = (_Float16*)(ws + 54525952);   // M*NKV
  _Float16* Vf  = (_Float16*)(ws + 58720256);   // M*NKV
  _Float16* Af  = (_Float16*)(ws + 62914560);   // M*NQ   (end ~76 MB)

  auto cvt = [&](const float* s, _Float16* d, int n) {
    int n4 = n / 4;
    cvt_f32_f16_v4<<<(n4 + 255) / 256, 256, 0, stream>>>((const float4*)s, d, n4);
  };
  cvt(x,  xh,  M_ * D_);
  cvt(Wq, Wqh, D_ * NQ_);
  cvt(Wk, Wkh, D_ * NKV_);
  cvt(Wv, Wvh, D_ * NKV_);
  cvt(Wo, Woh, NQ_ * D_);

  gemm_f16_wmma<false><<<dim3(NQ_ / 128,  M_ / 128), 256, 0, stream>>>(xh, Wqh, bq, Qf, NQ_,  D_);
  gemm_f16_wmma<false><<<dim3(NKV_ / 128, M_ / 128), 256, 0, stream>>>(xh, Wkh, bk, Kf, NKV_, D_);
  gemm_f16_wmma<false><<<dim3(NKV_ / 128, M_ / 128), 256, 0, stream>>>(xh, Wvh, bv, Vf, NKV_, D_);

  flash_gqa_wmma<<<dim3(L_ / 128, H_, B_), 256, 0, stream>>>(Qf, Kf, Vf, Af);

  gemm_f16_wmma<true><<<dim3(D_ / 128, M_ / 128), 256, 0, stream>>>(Af, Woh, bo, d_out, D_, NQ_);
}
